// Model_17085379903579
// MI455X (gfx1250) — matrix operand order repaired and checked by reading, not compiled, not run on hardware
//
#include <hip/hip_runtime.h>
#include <math.h>

// ---------------- types ----------------
typedef __bf16 bf16;
typedef __attribute__((ext_vector_type(16))) __bf16 v16bf;
typedef __attribute__((ext_vector_type(8)))  __bf16 v8bf;
typedef __attribute__((ext_vector_type(8)))  float  v8f;

union V16U { v16bf v; v8bf h[2]; };

// ---------------- model dims ----------------
#define DM     512
#define NB     16          // batch
#define LSEQ   512
#define NH     8
#define HD     64
#define DFF    2048
#define MROWS  (NB*LSEQ)   // 8192
#define PREDL  256
#define COUT   7
#define NMARK  4
#define CIN    7

// =====================================================================
// Batched WMMA GEMM:  C[M,N] = A[M,K] * Bt[N,K]^T  (Bt stored row-major [N,K])
// Block: 256 thr = 8 waves; wave w -> m-tile (w&3), n-group (w>>2) of 64.
// Block tile = 64(M) x 128(N). K stepped by 32 (bf16 16x16x32 WMMA).
// Out-of-range output columns are handled by CLAMPING the Bt row offset
// (valid memory, garbage accumulated, never stored) so the K-loop runs with
// full EXEC and zero per-iteration control flow. A and B fragments are
// addressed as base + invariant offset + kk so the compiler emits
// global_load_b128 with immediate offsets (no flat path / DScnt coupling).
// EPI: 0=none 1=+bias 2=+bias+gelu(exact) 3=+bias+residual
// OUT: 0=f32  1=bf16
// =====================================================================
struct GP {
  const bf16* A; const bf16* Bt; void* C;
  const float* bias; const float* Res;
  int lda, ldb, ldc, ldr;
  long sA, sA2, sB, sB2, sC, sC2, sR, sR2;
  int M, N, K, inner;
};

template<int EPI, int OUT>
__global__ __launch_bounds__(256) void gemm_wmma(GP p)
{
  const int bo = blockIdx.z / p.inner;
  const int bi = blockIdx.z - bo * p.inner;
  const bf16* __restrict__ A  = p.A  + (long)bo * p.sA + (long)bi * p.sA2;
  const bf16* __restrict__ Bt = p.Bt + (long)bo * p.sB + (long)bi * p.sB2;
  float* Cf = (float*)p.C + (long)bo * p.sC + (long)bi * p.sC2;
  bf16*  Cb = (bf16*) p.C + (long)bo * p.sC + (long)bi * p.sC2;
  const float* Res = p.Res ? (p.Res + (long)bo * p.sR + (long)bi * p.sR2) : nullptr;

  const int lane = threadIdx.x & 31;
  const int wave = threadIdx.x >> 5;
  const int mt = wave & 3;
  const int ng = wave >> 2;
  const int m0 = blockIdx.y * 64 + mt * 16;
  const int n0 = blockIdx.x * 128 + ng * 64;
  const bool hi = lane >= 16;
  const int ln16 = lane & 15;

  // A fragment base (16x32 bf16, contiguous layout: lanes0-15 K=0..15, hi +16)
  const int mrow = m0 + ln16;                 // M is a multiple of 64 here
  const bf16* ap = A + (long)mrow * p.lda + (hi ? 16 : 0);

  // B fragment element offsets from Bt, row-clamped (loop-invariant)
  long boff0, boff1, boff2, boff3;
  {
    const long kadd = hi ? 16 : 0;
    int n, nc;
    n = n0 + 0 * 16 + ln16; nc = (n < p.N) ? n : (p.N - 1); boff0 = (long)nc * p.ldb + kadd;
    n = n0 + 1 * 16 + ln16; nc = (n < p.N) ? n : (p.N - 1); boff1 = (long)nc * p.ldb + kadd;
    n = n0 + 2 * 16 + ln16; nc = (n < p.N) ? n : (p.N - 1); boff2 = (long)nc * p.ldb + kadd;
    n = n0 + 3 * 16 + ln16; nc = (n < p.N) ? n : (p.N - 1); boff3 = (long)nc * p.ldb + kadd;
  }

  v8f acc[4] = {};                            // 4 n-tiles of 16x16 f32

  for (int kk = 0; kk < p.K; kk += 32) {
    V16U a;
    a.h[0] = *(const v8bf*)(ap + kk);
    a.h[1] = *(const v8bf*)(ap + kk + 8);
    __builtin_prefetch(ap + kk + 32, 0, 3);   // global_prefetch next K chunk
    const v16bf b0 = *(const v16bf*)(Bt + boff0 + kk);
    acc[0] = __builtin_amdgcn_wmma_f32_16x16x32_bf16(
        false, a.v, false, b0, (short)0, acc[0], false, false);
    const v16bf b1 = *(const v16bf*)(Bt + boff1 + kk);
    acc[1] = __builtin_amdgcn_wmma_f32_16x16x32_bf16(
        false, a.v, false, b1, (short)0, acc[1], false, false);
    const v16bf b2 = *(const v16bf*)(Bt + boff2 + kk);
    acc[2] = __builtin_amdgcn_wmma_f32_16x16x32_bf16(
        false, a.v, false, b2, (short)0, acc[2], false, false);
    const v16bf b3 = *(const v16bf*)(Bt + boff3 + kk);
    acc[3] = __builtin_amdgcn_wmma_f32_16x16x32_bf16(
        false, a.v, false, b3, (short)0, acc[3], false, false);
  }

  // epilogue
#pragma unroll
  for (int t = 0; t < 4; ++t) {
    const int n = n0 + t * 16 + ln16;
    if (n >= p.N) continue;
    const float bv = (EPI >= 1) ? p.bias[n] : 0.0f;
#pragma unroll
    for (int r = 0; r < 8; ++r) {
      const int m = m0 + r + (hi ? 8 : 0);
      float v = acc[t][r] + bv;
      if (EPI == 2) v = 0.5f * v * (1.0f + erff(v * 0.70710678118654752f));
      if (EPI == 3) v += Res[(long)m * p.ldr + n];
      if (OUT == 0) Cf[(long)m * p.ldc + n] = v;
      else          Cb[(long)m * p.ldc + n] = (bf16)v;
    }
  }
}

template<int EPI, int OUT>
static void run_gemm(hipStream_t st, const GP& p, int nb) {
  dim3 grid((p.N + 127) / 128, (p.M + 63) / 64, nb);
  gemm_wmma<EPI, OUT><<<grid, dim3(256), 0, st>>>(p);
}

// =====================================================================
// LogSparse mask (faithful port, mask==1 -> -inf). One thread per row.
// =====================================================================
__global__ void build_mask(unsigned char* __restrict__ mask)
{
  const int i = blockIdx.x * blockDim.x + threadIdx.x;
  if (i >= LSEQ) return;
  unsigned char* row = mask + (long)i * LSEQ;
  for (int j = 0; j < LSEQ; ++j) row[j] = 0;
  const int log_l = 9, sub_len = LSEQ, win_len = LSEQ;
  if ((win_len / sub_len) * 2 * log_l > i) {
    for (int j = 0; j <= i; ++j) row[j] = 1;
  } else {
    int idx = i;
    while (idx >= 0) {
      if (idx - log_l + 1 < 0) { for (int j = 0; j < idx; ++j) row[j] = 1; break; }
      for (int j = idx - log_l + 1; j <= idx; ++j) row[j] = 1;
      for (int j = 0; j < log_l; ++j) {
        int ni = idx - log_l + 1 - (1 << j);
        if ((idx - ni) <= sub_len && ni >= 0) row[ni] = 1;
      }
      idx -= sub_len;
    }
  }
}

// =====================================================================
// Embedding: circular conv1d(k=3) + time-mark matmul + sinusoidal PE
// =====================================================================
__global__ __launch_bounds__(256) void embed_k(
    const float* __restrict__ x, const float* __restrict__ xmark,
    const float* __restrict__ convw, const float* __restrict__ timew,
    float* __restrict__ out)
{
  const long i = (long)blockIdx.x * 256 + threadIdx.x;
  if (i >= (long)NB * LSEQ * DM) return;
  const int d = (int)(i & (DM - 1));
  const long bl = i >> 9;
  const int l = (int)(bl & (LSEQ - 1));
  const int b = (int)(bl >> 9);
  float acc = 0.0f;
#pragma unroll
  for (int k = 0; k < 3; ++k) {
    int ls = l + k - 1; ls = (ls + LSEQ) & (LSEQ - 1);     // wrap pad
    const float* xr = x + ((long)b * LSEQ + ls) * CIN;
    const float* wr = convw + (long)k * CIN * DM + d;
#pragma unroll
    for (int c = 0; c < CIN; ++c) acc += xr[c] * wr[(long)c * DM];
  }
  const float* mr = xmark + ((long)b * LSEQ + l) * NMARK;
#pragma unroll
  for (int m = 0; m < NMARK; ++m) acc += mr[m] * timew[m * DM + d];
  const int i2 = d & ~1;
  const float div = __expf(-(float)i2 * (9.210340371976184f / (float)DM));
  const float ang = (float)l * div;
  acc += (d & 1) ? cosf(ang) : sinf(ang);
  out[i] = acc;
}

// ---------------- elementwise / transpose helpers ----------------
__global__ __launch_bounds__(256) void cvt_f2b(const float* __restrict__ X,
                                               bf16* __restrict__ Y, long n) {
  const long i = (long)blockIdx.x * 256 + threadIdx.x;
  if (i < n) Y[i] = (bf16)X[i];
}

// W[K,N] f32 -> Wt[N,K] bf16
__global__ __launch_bounds__(256) void wconv_t(const float* __restrict__ W,
                                               bf16* __restrict__ Wt, int K, int N) {
  const long i = (long)blockIdx.x * 256 + threadIdx.x;
  if (i >= (long)K * N) return;
  const int k = (int)(i / N), n = (int)(i % N);
  Wt[(long)n * K + k] = (bf16)W[i];
}

// V bf16 [B,S,H*64] -> Vt bf16 [B,H,64,S]
__global__ __launch_bounds__(256) void v_transpose(const bf16* __restrict__ V,
                                                   bf16* __restrict__ Vt) {
  const long i = (long)blockIdx.x * 256 + threadIdx.x;
  if (i >= (long)NB * LSEQ * DM) return;
  const int d = (int)(i & (DM - 1));
  const long bs = i >> 9;
  const int s = (int)(bs & (LSEQ - 1));
  const int b = (int)(bs >> 9);
  const int h = d >> 6, e = d & 63;
  Vt[(((long)b * NH + h) * HD + e) * LSEQ + s] = V[i];
}

// =====================================================================
// Masked softmax over 512-wide rows; writes bf16 probabilities.
// grid = (Lq, B*H); block = 256 (2 elems/thread)
// =====================================================================
__global__ __launch_bounds__(256) void softmax_mask(
    const float* __restrict__ S, bf16* __restrict__ P,
    const unsigned char* __restrict__ mask)
{
  const long row = (long)blockIdx.y * LSEQ + blockIdx.x;
  const float* srow = S + row * LSEQ;
  bf16* prow = P + row * LSEQ;
  const unsigned char* mrow = mask + (long)blockIdx.x * LSEQ;
  const int t = threadIdx.x;
  const float NEG = -__builtin_inff();
  float v0 = mrow[t]       ? NEG : srow[t]       * 0.125f;   // 1/sqrt(64)
  float v1 = mrow[t + 256] ? NEG : srow[t + 256] * 0.125f;
  float mx = fmaxf(v0, v1);
  for (int o = 16; o; o >>= 1) mx = fmaxf(mx, __shfl_xor(mx, o, 32));
  __shared__ float red[8];
  if ((t & 31) == 0) red[t >> 5] = mx;
  __syncthreads();
  float gmax = red[0];
#pragma unroll
  for (int w = 1; w < 8; ++w) gmax = fmaxf(gmax, red[w]);
  __syncthreads();
  const float e0 = __expf(v0 - gmax);
  const float e1 = __expf(v1 - gmax);
  float s = e0 + e1;
  for (int o = 16; o; o >>= 1) s += __shfl_xor(s, o, 32);
  if ((t & 31) == 0) red[t >> 5] = s;
  __syncthreads();
  float tot = 0.0f;
#pragma unroll
  for (int w = 0; w < 8; ++w) tot += red[w];
  const float rs = 1.0f / tot;
  prow[t]       = (bf16)(e0 * rs);
  prow[t + 256] = (bf16)(e1 * rs);
}

// =====================================================================
// LayerNorm over 512 cols; writes f32 + bf16. grid = MROWS, block = 256
// =====================================================================
__global__ __launch_bounds__(256) void layernorm_k(
    const float* __restrict__ X, const float* __restrict__ g,
    const float* __restrict__ b, float* __restrict__ Yf, bf16* __restrict__ Yb)
{
  const long row = blockIdx.x;
  const float* xr = X + row * DM;
  const int t = threadIdx.x;
  const float x0 = xr[t], x1 = xr[t + 256];
  float s = x0 + x1, s2 = x0 * x0 + x1 * x1;
  for (int o = 16; o; o >>= 1) { s += __shfl_xor(s, o, 32); s2 += __shfl_xor(s2, o, 32); }
  __shared__ float r1[8], r2[8];
  if ((t & 31) == 0) { r1[t >> 5] = s; r2[t >> 5] = s2; }
  __syncthreads();
  float ts = 0.0f, ts2 = 0.0f;
#pragma unroll
  for (int w = 0; w < 8; ++w) { ts += r1[w]; ts2 += r2[w]; }
  const float mean = ts * (1.0f / DM);
  const float var = ts2 * (1.0f / DM) - mean * mean;
  const float inv = rsqrtf(var + 1e-5f);
  const float y0 = (x0 - mean) * inv * g[t] + b[t];
  const float y1 = (x1 - mean) * inv * g[t + 256] + b[t + 256];
  Yf[row * DM + t] = y0;       Yf[row * DM + t + 256] = y1;
  Yb[row * DM + t] = (bf16)y0; Yb[row * DM + t + 256] = (bf16)y1;
}

// full [B,512,7] -> out [B,256,7] (last PRED_LEN)
__global__ __launch_bounds__(256) void gather_out(const float* __restrict__ full,
                                                  float* __restrict__ out) {
  const int i = blockIdx.x * 256 + threadIdx.x;
  if (i >= NB * PREDL * COUT) return;
  const int c = i % COUT;
  const int tpos = (i / COUT) % PREDL;
  const int b = i / (PREDL * COUT);
  out[i] = full[((long)b * LSEQ + (LSEQ - PREDL) + tpos) * COUT + c];
}

// =====================================================================
// Host orchestration
// =====================================================================
extern "C" void kernel_launch(void* const* d_in, const int* in_sizes, int n_in,
                              void* d_out, int out_size, void* d_ws, size_t ws_size,
                              hipStream_t stream)
{
  (void)in_sizes; (void)n_in; (void)out_size; (void)ws_size;
  auto F = [&](int i) -> const float* { return (const float*)d_in[i]; };

  // ---- workspace bump allocator ----
  char* wsb = (char*)d_ws;
  size_t off = 0;
  auto alloc = [&](size_t bytes) -> void* {
    off = (off + 255) & ~(size_t)255;
    void* p = wsb + off; off += bytes; return p;
  };

  const size_t SZ_ACTF  = (size_t)MROWS * DM * 4;
  const size_t SZ_ACTB  = (size_t)MROWS * DM * 2;
  const size_t SZ_SCORE = (size_t)NB * NH * LSEQ * LSEQ * 4;   // 134MB

  unsigned char* mask = (unsigned char*)alloc((size_t)LSEQ * LSEQ);
  float* actF[2] = { (float*)alloc(SZ_ACTF), (float*)alloc(SZ_ACTF) };
  bf16*  actB[2] = { (bf16*) alloc(SZ_ACTB), (bf16*) alloc(SZ_ACTB) };
  float* encF = (float*)alloc(SZ_ACTF);
  bf16*  encB = (bf16*) alloc(SZ_ACTB);
  bf16* Qb = (bf16*)alloc(SZ_ACTB);
  bf16* Kb = (bf16*)alloc(SZ_ACTB);
  bf16* Vb = (bf16*)alloc(SZ_ACTB);
  bf16* Vt = (bf16*)alloc(SZ_ACTB);
  float* big = (float*)alloc(SZ_SCORE);       // scores; aliased below
  float* scoresF = big;
  float* Yf = big;                                        // dead-scores alias
  bf16*  Hb = (bf16*)((char*)big + ((size_t)64 << 20));   // FFN hidden alias
  bf16* Pb = (bf16*)alloc(SZ_SCORE / 2);                  // bf16 probs
  bf16* AVb = (bf16*)alloc(SZ_ACTB);
  float* projFull = (float*)alloc((size_t)MROWS * COUT * 4);

  // ---- weights: f32 [K,N] -> bf16 [N,K] ----
  auto wt = [&](const float* W, int K, int N) -> bf16* {
    bf16* o = (bf16*)alloc((size_t)K * N * 2);
    long n = (long)K * N;
    wconv_t<<<(unsigned)((n + 255) / 256), 256, 0, stream>>>(W, o, K, N);
    return o;
  };

  struct AttnT { bf16 *wqt,*wkt,*wvt,*wot; const float *bq,*bk,*bv,*bo; };
  auto get_attn = [&](int base) {
    AttnT a;
    a.wqt = wt(F(base+0), DM, DM); a.bq = F(base+1);
    a.wkt = wt(F(base+2), DM, DM); a.bk = F(base+3);
    a.wvt = wt(F(base+4), DM, DM); a.bv = F(base+5);
    a.wot = wt(F(base+6), DM, DM); a.bo = F(base+7);
    return a;
  };

  // param layout (setup_inputs insertion order): 0..3 data tensors, then params
  AttnT encA[2]; bf16 *encW1t[2], *encW2t[2];
  const float *encB1f[2], *encB2f[2], *encG1[2], *encBb1[2], *encG2[2], *encBb2[2];
  for (int l = 0; l < 2; ++l) {
    const int base = 8 + l * 16;
    encA[l] = get_attn(base);
    encW1t[l] = wt(F(base+8), DM, DFF);  encB1f[l] = F(base+9);
    encW2t[l] = wt(F(base+10), DFF, DM); encB2f[l] = F(base+11);
    encG1[l] = F(base+12); encBb1[l] = F(base+13);
    encG2[l] = F(base+14); encBb2[l] = F(base+15);
  }
  const float *encNg = F(40), *encNb = F(41);
  AttnT decS = get_attn(42);
  AttnT decC = get_attn(50);
  bf16* decW1t = wt(F(58), DM, DFF);  const float* decB1f = F(59);
  bf16* decW2t = wt(F(60), DFF, DM);  const float* decB2f = F(61);
  const float *dG1=F(62),*dB1=F(63),*dG2=F(64),*dB2=F(65),*dG3=F(66),*dB3=F(67);
  const float *decNg = F(68), *decNb = F(69);
  bf16* projWt = wt(F(70), DM, COUT); const float* projB = F(71);

  build_mask<<<2, 256, 0, stream>>>(mask);

  const long nAct = (long)MROWS * DM;
  auto cvt = [&](const float* x, bf16* y) {
    cvt_f2b<<<(unsigned)((nAct + 255) / 256), 256, 0, stream>>>(x, y, nAct);
  };
  auto ln = [&](const float* x, const float* g, const float* b, float* yf, bf16* yb) {
    layernorm_k<<<MROWS, 256, 0, stream>>>(x, g, b, yf, yb);
  };

  // ---- attention block: Yf = resF + O(attn(qB; kvB)) ----
  auto attn_block = [&](const float* resF, const bf16* qB, const bf16* kvB,
                        const AttnT& w, float* outYf) {
    GP p = {}; p.inner = 1; p.lda = DM; p.ldb = DM; p.ldc = DM;
    p.M = MROWS; p.N = DM; p.K = DM;
    p.A = qB;  p.Bt = w.wqt; p.C = Qb; p.bias = w.bq; run_gemm<1,1>(stream, p, 1);
    p.A = kvB; p.Bt = w.wkt; p.C = Kb; p.bias = w.bk; run_gemm<1,1>(stream, p, 1);
    p.A = kvB; p.Bt = w.wvt; p.C = Vb; p.bias = w.bv; run_gemm<1,1>(stream, p, 1);
    v_transpose<<<(unsigned)((nAct + 255) / 256), 256, 0, stream>>>(Vb, Vt);
    // scores[b,h,l,s] = Q . K  (batched 16x8; per-head K-dim = 64)
    GP s = {}; s.inner = NH; s.M = LSEQ; s.N = LSEQ; s.K = HD;
    s.A = Qb; s.lda = DM; s.sA = (long)LSEQ * DM; s.sA2 = HD;
    s.Bt = Kb; s.ldb = DM; s.sB = (long)LSEQ * DM; s.sB2 = HD;
    s.C = scoresF; s.ldc = LSEQ; s.sC = (long)NH * LSEQ * LSEQ; s.sC2 = (long)LSEQ * LSEQ;
    run_gemm<0,0>(stream, s, NB * NH);
    softmax_mask<<<dim3(LSEQ, NB * NH), 256, 0, stream>>>(scoresF, Pb, mask);
    // AV: [512,64] = P[512,512] x Vt[64,512]^T
    GP a = {}; a.inner = NH; a.M = LSEQ; a.N = HD; a.K = LSEQ;
    a.A = Pb; a.lda = LSEQ; a.sA = (long)NH * LSEQ * LSEQ; a.sA2 = (long)LSEQ * LSEQ;
    a.Bt = Vt; a.ldb = LSEQ; a.sB = (long)NH * HD * LSEQ; a.sB2 = (long)HD * LSEQ;
    a.C = AVb; a.ldc = DM; a.sC = (long)LSEQ * DM; a.sC2 = HD;
    run_gemm<0,1>(stream, a, NB * NH);
    // out projection + bias + residual -> f32
    GP o = {}; o.inner = 1; o.M = MROWS; o.N = DM; o.K = DM;
    o.A = AVb; o.lda = DM; o.Bt = w.wot; o.ldb = DM;
    o.C = outYf; o.ldc = DM; o.bias = w.bo; o.Res = resF; o.ldr = DM;
    run_gemm<3,0>(stream, o, 1);
  };

  auto ffn_block = [&](const float* inF, const bf16* inB,
                       const bf16* w1t, const float* b1f,
                       const bf16* w2t, const float* b2f, float* outYf) {
    GP p = {}; p.inner = 1; p.M = MROWS; p.N = DFF; p.K = DM;
    p.A = inB; p.lda = DM; p.Bt = w1t; p.ldb = DM; p.C = Hb; p.ldc = DFF; p.bias = b1f;
    run_gemm<2,1>(stream, p, 1);                     // GELU -> bf16 hidden
    GP q = {}; q.inner = 1; q.M = MROWS; q.N = DM; q.K = DFF;
    q.A = Hb; q.lda = DFF; q.Bt = w2t; q.ldb = DFF; q.C = outYf; q.ldc = DM;
    q.bias = b2f; q.Res = inF; q.ldr = DM;
    run_gemm<3,0>(stream, q, 1);                     // +residual -> f32
  };

  const unsigned embGrid = (unsigned)(((long)NB * LSEQ * DM + 255) / 256);

  // ================= ENCODER =================
  embed_k<<<embGrid, 256, 0, stream>>>(F(0), F(1), F(4), F(5), actF[0]);
  cvt(actF[0], actB[0]);
  for (int l = 0; l < 2; ++l) {
    attn_block(actF[0], actB[0], actB[0], encA[l], Yf);
    ln(Yf, encG1[l], encBb1[l], actF[1], actB[1]);
    ffn_block(actF[1], actB[1], encW1t[l], encB1f[l], encW2t[l], encB2f[l], Yf);
    ln(Yf, encG2[l], encBb2[l], actF[0], actB[0]);
  }
  ln(actF[0], encNg, encNb, encF, encB);             // encoder output

  // ================= DECODER =================
  embed_k<<<embGrid, 256, 0, stream>>>(F(2), F(3), F(6), F(7), actF[0]);
  cvt(actF[0], actB[0]);
  attn_block(actF[0], actB[0], actB[0], decS, Yf);   // self attn
  ln(Yf, dG1, dB1, actF[1], actB[1]);
  attn_block(actF[1], actB[1], encB, decC, Yf);      // cross attn (KV from encoder)
  ln(Yf, dG2, dB2, actF[0], actB[0]);
  ffn_block(actF[0], actB[0], decW1t, decB1f, decW2t, decB2f, Yf);
  ln(Yf, dG3, dB3, actF[1], actB[1]);
  ln(actF[1], decNg, decNb, actF[0], actB[0]);       // dec_norm

  // final projection [8192,512] x [512,7] + b
  GP pr = {}; pr.inner = 1; pr.M = MROWS; pr.N = COUT; pr.K = DM;
  pr.A = actB[0]; pr.lda = DM; pr.Bt = projWt; pr.ldb = DM;
  pr.C = projFull; pr.ldc = COUT; pr.bias = projB;
  run_gemm<1,0>(stream, pr, 1);

  gather_out<<<(NB * PREDL * COUT + 255) / 256, 256, 0, stream>>>(projFull, (float*)d_out);
}
